// CodeGNN_56307021250752
// MI455X (gfx1250) — compile-verified, matrix operand order checked by reading, not run on hardware
//
#include <hip/hip_runtime.h>

// ---------------------------------------------------------------------------
// CodeGNN on MI455X (gfx1250, wave32).
//  - All dense projections via v_wmma_f32_16x16x32_bf16 (f32 accum).
//  - Node-side factoring of edge GEMMs: k = k_node[src] + k_edge[e].
//  - Edge softmax via ordered-uint atomicMax + f32 atomicAdd (3 edge passes).
// ---------------------------------------------------------------------------

typedef __attribute__((ext_vector_type(16))) __bf16 v16bf;
typedef __attribute__((ext_vector_type(8)))  __bf16 v8bf;
typedef __attribute__((ext_vector_type(8)))  float  v8f;

static constexpr int   N_  = 20000;
static constexpr int   E_  = 640000;
static constexpr int   IN_ = 64;
static constexpr int   H_  = 128;
static constexpr int   G_  = 64;

#define DEVI __device__ __forceinline__

DEVI __bf16 f2bf(float f) {
    unsigned u = __float_as_uint(f);
    unsigned r = u + 0x7fffu + ((u >> 16) & 1u);   // round-to-nearest-even
    union { unsigned short s; __bf16 b; } cv;
    cv.s = (unsigned short)(r >> 16);
    return cv.b;
}
DEVI unsigned f2o(float f) {  // monotonic float->uint for atomicMax
    unsigned u = __float_as_uint(f);
    return (u >> 31) ? ~u : (u | 0x80000000u);
}
DEVI float o2f(unsigned s) {
    unsigned u = (s >> 31) ? (s & 0x7fffffffu) : ~s;
    return __uint_as_float(u);
}

// ------------------------------- utility kernels ---------------------------
__global__ void k_fill_u32(unsigned* __restrict__ p, unsigned v, long n) {
    long i = (long)blockIdx.x * blockDim.x + threadIdx.x;
    if (i < n) p[i] = v;
}

__global__ void k_cast_bf16(const float* __restrict__ in, __bf16* __restrict__ out, long n) {
    long i = (long)blockIdx.x * blockDim.x + threadIdx.x;
    if (i < n) out[i] = f2bf(in[i]);
}

// W[K][H] f32 -> Wt[H][K] bf16  (B-operand, row-per-output-channel)
__global__ void k_transpose_wt(const float* __restrict__ W, __bf16* __restrict__ Wt, int K) {
    long i = (long)blockIdx.x * blockDim.x + threadIdx.x;
    if (i >= (long)K * H_) return;
    int k = (int)(i % K);
    int h = (int)(i / K);
    Wt[(long)h * K + k] = f2bf(W[(long)k * H_ + h]);
}

// ------------------------------- WMMA GEMM ---------------------------------
struct GemmSegs {
    const __bf16* A[4];
    int lda[4];
    int koff[4];
    int K[4];
    int nseg;
};

// C[M][128] = sum_seg A_seg @ Wt[:, koff:koff+K]^T  (+ bias)
// block = 256 threads = 8 waves; wave w -> 16x16 output tile, cols w*16..
__global__ __launch_bounds__(256)
void k_wmma_gemm(GemmSegs segs, const __bf16* __restrict__ Wt, int Ktot,
                 const float* __restrict__ bias, float* __restrict__ C, int M)
{
    const int lane = threadIdx.x & 31;
    const int wv   = threadIdx.x >> 5;     // 0..7 column tile
    const int rt   = blockIdx.x;           // row tile
    const int l15  = lane & 15;
    const int hi   = lane >> 4;
    const int kb   = hi * 8;               // ISA 16-bit A/B layout: K 0-7/16-23 (lo half), 8-15/24-31 (hi half)
    int mrow = rt * 16 + l15;
    if (mrow >= M) mrow = M - 1;           // clamp (stores are guarded)
    const int col = wv * 16 + l15;

    v8f acc = {0.f, 0.f, 0.f, 0.f, 0.f, 0.f, 0.f, 0.f};
    const __bf16* wrowBase = Wt + (long)col * Ktot;

    for (int s = 0; s < segs.nseg; ++s) {
        const __bf16* arow = segs.A[s] + (long)mrow * segs.lda[s];
        const __bf16* wrow = wrowBase + segs.koff[s];
        const int Ks = segs.K[s];
        for (int k0 = 0; k0 < Ks; k0 += 32) {
            v8bf alo = *(const v8bf*)(arow + k0 + kb);
            v8bf ahi = *(const v8bf*)(arow + k0 + 16 + kb);
            v8bf blo = *(const v8bf*)(wrow + k0 + kb);
            v8bf bhi = *(const v8bf*)(wrow + k0 + 16 + kb);
            v16bf a = __builtin_shufflevector(alo, ahi, 0,1,2,3,4,5,6,7,8,9,10,11,12,13,14,15);
            v16bf b = __builtin_shufflevector(blo, bhi, 0,1,2,3,4,5,6,7,8,9,10,11,12,13,14,15);
            acc = __builtin_amdgcn_wmma_f32_16x16x32_bf16(false, a, false, b,
                                                          (short)0, acc, false, false);
        }
    }
    const float bv = bias ? bias[col] : 0.f;
#pragma unroll
    for (int r = 0; r < 8; ++r) {          // C/D layout: vgpr r, lanes16-31 -> M+8
        const int row = rt * 16 + r + hi * 8;
        if (row < M) C[(long)row * H_ + col] = acc[r] + bv;
    }
}

// ------------------------------- edge softmax ------------------------------
__global__ __launch_bounds__(256)
void k_edge_max(const int* __restrict__ src, const int* __restrict__ dst,
                const float* __restrict__ q, const float* __restrict__ kn,
                const float* __restrict__ ke, unsigned* __restrict__ mord)
{
    long t = (long)blockIdx.x * blockDim.x + threadIdx.x;
    if (t >= (long)E_ * 32) return;
    const int e = (int)(t >> 5);
    const int c = ((int)t & 31) * 4;
    const int sN = src[e], dN = dst[e];
    float4 q4 = *(const float4*)(q  + (long)dN * H_ + c);
    float4 n4 = *(const float4*)(kn + (long)sN * H_ + c);
    float4 e4 = *(const float4*)(ke + (long)e  * H_ + c);
    unsigned* mp = mord + (long)dN * H_ + c;
    atomicMax(mp + 0, f2o(q4.x * (n4.x + e4.x)));
    atomicMax(mp + 1, f2o(q4.y * (n4.y + e4.y)));
    atomicMax(mp + 2, f2o(q4.z * (n4.z + e4.z)));
    atomicMax(mp + 3, f2o(q4.w * (n4.w + e4.w)));
}

__global__ void k_decode_max(unsigned* __restrict__ m, long n) {
    long i = (long)blockIdx.x * blockDim.x + threadIdx.x;
    if (i >= n) return;
    float v = o2f(m[i]);
    if (!isfinite(v)) v = 0.f;             // reference: where(isfinite(m), m, 0)
    ((float*)m)[i] = v;
}

__global__ __launch_bounds__(256)
void k_edge_expsum(const int* __restrict__ src, const int* __restrict__ dst,
                   const float* __restrict__ q, const float* __restrict__ kn,
                   const float* __restrict__ ke, const float* __restrict__ m,
                   float* __restrict__ ebuf, float* __restrict__ ssum)
{
    long t = (long)blockIdx.x * blockDim.x + threadIdx.x;
    if (t >= (long)E_ * 32) return;
    const int e = (int)(t >> 5);
    const int c = ((int)t & 31) * 4;
    const int sN = src[e], dN = dst[e];
    float4 q4 = *(const float4*)(q  + (long)dN * H_ + c);
    float4 n4 = *(const float4*)(kn + (long)sN * H_ + c);
    float4 k4 = *(const float4*)(ke + (long)e  * H_ + c);
    float4 m4 = *(const float4*)(m  + (long)dN * H_ + c);
    float4 ex;
    ex.x = __expf(q4.x * (n4.x + k4.x) - m4.x);
    ex.y = __expf(q4.y * (n4.y + k4.y) - m4.y);
    ex.z = __expf(q4.z * (n4.z + k4.z) - m4.z);
    ex.w = __expf(q4.w * (n4.w + k4.w) - m4.w);
    *(float4*)(ebuf + (long)e * H_ + c) = ex;
    float* sp = ssum + (long)dN * H_ + c;
    atomicAdd(sp + 0, ex.x); atomicAdd(sp + 1, ex.y);
    atomicAdd(sp + 2, ex.z); atomicAdd(sp + 3, ex.w);
}

__global__ __launch_bounds__(256)
void k_edge_aggr(const int* __restrict__ src, const int* __restrict__ dst,
                 const float* __restrict__ ebuf, const float* __restrict__ ssum,
                 const float* __restrict__ vn, const float* __restrict__ ve,
                 float* __restrict__ hn)
{
    long t = (long)blockIdx.x * blockDim.x + threadIdx.x;
    if (t >= (long)E_ * 32) return;
    const int e = (int)(t >> 5);
    const int c = ((int)t & 31) * 4;
    const int sN = src[e], dN = dst[e];
    float4 e4 = *(const float4*)(ebuf + (long)e  * H_ + c);
    float4 s4 = *(const float4*)(ssum + (long)dN * H_ + c);
    float4 n4 = *(const float4*)(vn   + (long)sN * H_ + c);
    float4 v4 = *(const float4*)(ve   + (long)e  * H_ + c);
    float* hp = hn + (long)dN * H_ + c;
    atomicAdd(hp + 0, (e4.x / s4.x) * (n4.x + v4.x));
    atomicAdd(hp + 1, (e4.y / s4.y) * (n4.y + v4.y));
    atomicAdd(hp + 2, (e4.z / s4.z) * (n4.z + v4.z));
    atomicAdd(hp + 3, (e4.w / s4.w) * (n4.w + v4.w));
}

// ------------------------------- layernorm ---------------------------------
__global__ __launch_bounds__(256)
void k_layernorm(const float* __restrict__ in, const float* __restrict__ g,
                 const float* __restrict__ b, float* __restrict__ outf,
                 __bf16* __restrict__ outb, int M)
{
    const int lane = threadIdx.x & 31;
    const int row  = blockIdx.x * 8 + (threadIdx.x >> 5);
    if (row >= M) return;
    const int c = lane * 4;
    float4 x = *(const float4*)(in + (long)row * H_ + c);
    float s = x.x + x.y + x.z + x.w;
#pragma unroll
    for (int m = 16; m >= 1; m >>= 1) s += __shfl_xor(s, m, 32);
    const float mean = s * (1.0f / H_);
    float d0 = x.x - mean, d1 = x.y - mean, d2 = x.z - mean, d3 = x.w - mean;
    float vs = d0 * d0 + d1 * d1 + d2 * d2 + d3 * d3;
#pragma unroll
    for (int m = 16; m >= 1; m >>= 1) vs += __shfl_xor(vs, m, 32);
    const float rstd = rsqrtf(vs * (1.0f / H_) + 1e-5f);
    float4 gg = *(const float4*)(g + c);
    float4 bb = *(const float4*)(b + c);
    float4 y;
    y.x = d0 * rstd * gg.x + bb.x;
    y.y = d1 * rstd * gg.y + bb.y;
    y.z = d2 * rstd * gg.z + bb.z;
    y.w = d3 * rstd * gg.w + bb.w;
    *(float4*)(outf + (long)row * H_ + c) = y;
    if (outb) {
        __bf16* p = outb + (long)row * H_ + c;
        p[0] = f2bf(y.x); p[1] = f2bf(y.y); p[2] = f2bf(y.z); p[3] = f2bf(y.w);
    }
}

// ------------------------------- readout -----------------------------------
__global__ __launch_bounds__(256)
void k_gate_logit(const float* __restrict__ h1, const float* __restrict__ gw,
                  const float* __restrict__ gb, const int* __restrict__ gid,
                  float* __restrict__ logit, unsigned* __restrict__ gmord, int M)
{
    const int lane = threadIdx.x & 31;
    const int row  = blockIdx.x * 8 + (threadIdx.x >> 5);
    if (row >= M) return;
    const int c = lane * 4;
    float4 x = *(const float4*)(h1 + (long)row * H_ + c);
    float4 w = *(const float4*)(gw + c);
    float p = x.x * w.x + x.y * w.y + x.z * w.z + x.w * w.w;
#pragma unroll
    for (int m = 16; m >= 1; m >>= 1) p += __shfl_xor(p, m, 32);
    if (lane == 0) {
        float L = p + gb[0];
        logit[row] = L;
        atomicMax(gmord + gid[row], f2o(L));
    }
}

__global__ void k_gate_exp(const float* __restrict__ logit, const float* __restrict__ gm,
                           const int* __restrict__ gid, float* __restrict__ ge,
                           float* __restrict__ gs, int M)
{
    int i = blockIdx.x * blockDim.x + threadIdx.x;
    if (i >= M) return;
    float v = __expf(logit[i] - gm[gid[i]]);
    ge[i] = v;
    atomicAdd(gs + gid[i], v);
}

__global__ __launch_bounds__(256)
void k_gate_out(const float* __restrict__ h1, const float* __restrict__ ge,
                const float* __restrict__ gs, const int* __restrict__ gid,
                float* __restrict__ out, int M)
{
    const int lane = threadIdx.x & 31;
    const int row  = blockIdx.x * 8 + (threadIdx.x >> 5);
    if (row >= M) return;
    const int g = gid[row];
    const float gate = ge[row] / gs[g];
    const float* hp = h1 + (long)row * H_;
    float* op = out + (long)g * H_;
#pragma unroll
    for (int c = lane; c < H_; c += 32) atomicAdd(op + c, gate * hp[c]);
}

// ------------------------------- host side ---------------------------------
static GemmSegs seg1(const __bf16* a, int l, int o, int k) {
    GemmSegs s{}; s.A[0]=a; s.lda[0]=l; s.koff[0]=o; s.K[0]=k; s.nseg=1; return s;
}
static GemmSegs seg2(const __bf16* a0,int l0,int o0,int k0,
                     const __bf16* a1,int l1,int o1,int k1) {
    GemmSegs s = seg1(a0,l0,o0,k0);
    s.A[1]=a1; s.lda[1]=l1; s.koff[1]=o1; s.K[1]=k1; s.nseg=2; return s;
}
static GemmSegs seg3(const __bf16* a0,int l0,int o0,int k0,
                     const __bf16* a1,int l1,int o1,int k1,
                     const __bf16* a2,int l2,int o2,int k2) {
    GemmSegs s = seg2(a0,l0,o0,k0,a1,l1,o1,k1);
    s.A[2]=a2; s.lda[2]=l2; s.koff[2]=o2; s.K[2]=k2; s.nseg=3; return s;
}
static GemmSegs seg4(const __bf16* a0,int l0,int o0,int k0,
                     const __bf16* a1,int l1,int o1,int k1,
                     const __bf16* a2,int l2,int o2,int k2,
                     const __bf16* a3,int l3,int o3,int k3) {
    GemmSegs s = seg3(a0,l0,o0,k0,a1,l1,o1,k1,a2,l2,o2,k2);
    s.A[3]=a3; s.lda[3]=l3; s.koff[3]=o3; s.K[3]=k3; s.nseg=4; return s;
}

extern "C" void kernel_launch(void* const* d_in, const int* in_sizes, int n_in,
                              void* d_out, int out_size, void* d_ws, size_t ws_size,
                              hipStream_t stream)
{
    (void)in_sizes; (void)n_in; (void)out_size;
    const float* kind  = (const float*)d_in[0];
    const float* ntype = (const float*)d_in[1];
    const float* eh    = (const float*)d_in[2];
    const float* Qw  = (const float*)d_in[3];  const float* Qb  = (const float*)d_in[4];
    const float* Kw  = (const float*)d_in[5];  const float* Kb  = (const float*)d_in[6];
    const float* Vw  = (const float*)d_in[7];  const float* Vb  = (const float*)d_in[8];
    const float* Ww  = (const float*)d_in[9];  const float* Wb  = (const float*)d_in[10];
    const float* Q2w = (const float*)d_in[11]; const float* Q2b = (const float*)d_in[12];
    const float* K2w = (const float*)d_in[13]; const float* K2b = (const float*)d_in[14];
    const float* V2w = (const float*)d_in[15]; const float* V2b = (const float*)d_in[16];
    const float* W2w = (const float*)d_in[17]; const float* W2b = (const float*)d_in[18];
    const float* lng = (const float*)d_in[19]; const float* lnb = (const float*)d_in[20];
    const float* gw  = (const float*)d_in[21]; const float* gb  = (const float*)d_in[22];
    const int* src = (const int*)d_in[23];
    const int* dst = (const int*)d_in[24];
    const int* gid = (const int*)d_in[25];
    float* out = (float*)d_out;

    // ------- workspace layout (deterministic, 256B aligned) -------
    char* ws = (char*)d_ws;
    size_t cur = 0;
    auto alloc = [&](size_t bytes) -> char* {
        cur = (cur + 255) & ~(size_t)255;
        char* p = ws + cur;
        cur += bytes;
        return p;
    };
    __bf16* kindb  = (__bf16*)alloc((size_t)N_ * IN_ * 2);
    __bf16* ntypeb = (__bf16*)alloc((size_t)N_ * IN_ * 2);
    __bf16* ehb    = (__bf16*)alloc((size_t)E_ * IN_ * 2);
    __bf16* hb     = (__bf16*)alloc((size_t)N_ * H_ * 2);
    __bf16* aggb   = (__bf16*)alloc((size_t)N_ * H_ * 2);   // bf16 of h_n / h_n1
    __bf16* Qwt  = (__bf16*)alloc((size_t)128 * H_ * 2);
    __bf16* Kwt  = (__bf16*)alloc((size_t)192 * H_ * 2);
    __bf16* Vwt  = (__bf16*)alloc((size_t)192 * H_ * 2);
    __bf16* Wwt  = (__bf16*)alloc((size_t)256 * H_ * 2);
    __bf16* Q2wt = (__bf16*)alloc((size_t)256 * H_ * 2);
    __bf16* K2wt = (__bf16*)alloc((size_t)320 * H_ * 2);
    __bf16* V2wt = (__bf16*)alloc((size_t)320 * H_ * 2);
    __bf16* W2wt = (__bf16*)alloc((size_t)384 * H_ * 2);
    float* qbuf   = (float*)alloc((size_t)N_ * H_ * 4);   // q  / q2
    float* knbuf  = (float*)alloc((size_t)N_ * H_ * 4);   // kn / kn2
    float* vnbuf  = (float*)alloc((size_t)N_ * H_ * 4);   // vn / vn2
    float* mbuf   = (float*)alloc((size_t)N_ * H_ * 4);   // ordered max, then decoded
    float* sbuf   = (float*)alloc((size_t)N_ * H_ * 4);
    float* hnbuf  = (float*)alloc((size_t)N_ * H_ * 4);
    float* preln  = (float*)alloc((size_t)N_ * H_ * 4);
    float* hbuf   = (float*)alloc((size_t)N_ * H_ * 4);   // h (f32, unused after bf16 made, kept for clarity)
    float* h1buf  = (float*)alloc((size_t)N_ * H_ * 4);
    float* logitb = (float*)alloc((size_t)N_ * 4);
    float* gebuf  = (float*)alloc((size_t)N_ * 4);
    float* gmbuf  = (float*)alloc((size_t)G_ * 4);
    float* gsbuf  = (float*)alloc((size_t)G_ * 4);
    float* kebuf  = (float*)alloc((size_t)E_ * H_ * 4);   // ke / ke2
    float* vebuf  = (float*)alloc((size_t)E_ * H_ * 4);   // ve / ve2
    float* ebuf   = (float*)alloc((size_t)E_ * H_ * 4);   // softmax numerators
    if (cur > ws_size) return;   // workspace too small: bail deterministically

    auto blk = [](long n, int b) { return (unsigned)((n + b - 1) / b); };
    const unsigned ORD_NEG_INF = 0x007FFFFFu;   // f2o(-inf)
    const long NH = (long)N_ * H_;
    const long EW = (long)E_ * 32;               // edge pass threads

    // ------- bf16 casts + weight transposes -------
    k_cast_bf16<<<blk((long)N_*IN_,256),256,0,stream>>>(kind,  kindb,  (long)N_*IN_);
    k_cast_bf16<<<blk((long)N_*IN_,256),256,0,stream>>>(ntype, ntypeb, (long)N_*IN_);
    k_cast_bf16<<<blk((long)E_*IN_,256),256,0,stream>>>(eh,    ehb,    (long)E_*IN_);
    k_transpose_wt<<<blk((long)128*H_,256),256,0,stream>>>(Qw,  Qwt,  128);
    k_transpose_wt<<<blk((long)192*H_,256),256,0,stream>>>(Kw,  Kwt,  192);
    k_transpose_wt<<<blk((long)192*H_,256),256,0,stream>>>(Vw,  Vwt,  192);
    k_transpose_wt<<<blk((long)256*H_,256),256,0,stream>>>(Ww,  Wwt,  256);
    k_transpose_wt<<<blk((long)256*H_,256),256,0,stream>>>(Q2w, Q2wt, 256);
    k_transpose_wt<<<blk((long)320*H_,256),256,0,stream>>>(K2w, K2wt, 320);
    k_transpose_wt<<<blk((long)320*H_,256),256,0,stream>>>(V2w, V2wt, 320);
    k_transpose_wt<<<blk((long)384*H_,256),256,0,stream>>>(W2w, W2wt, 384);

    const unsigned gN = blk(N_, 16), gE = blk(E_, 16);

    // ------- layer 1 projections -------
    k_wmma_gemm<<<gN,256,0,stream>>>(seg2(kindb,64,0,64, ntypeb,64,64,64), Qwt, 128, Qb,   qbuf,  N_);
    k_wmma_gemm<<<gN,256,0,stream>>>(seg2(kindb,64,0,64, ntypeb,64,64,64), Kwt, 192, nullptr, knbuf, N_);
    k_wmma_gemm<<<gN,256,0,stream>>>(seg2(kindb,64,0,64, ntypeb,64,64,64), Vwt, 192, nullptr, vnbuf, N_);
    k_wmma_gemm<<<gE,256,0,stream>>>(seg1(ehb,64,128,64),                  Kwt, 192, Kb,   kebuf, E_);
    k_wmma_gemm<<<gE,256,0,stream>>>(seg1(ehb,64,128,64),                  Vwt, 192, Vb,   vebuf, E_);

    // ------- layer 1 edge softmax + aggregate -------
    k_fill_u32<<<blk(NH,256),256,0,stream>>>((unsigned*)mbuf, ORD_NEG_INF, NH);
    k_fill_u32<<<blk(NH,256),256,0,stream>>>((unsigned*)sbuf, 0u, NH);
    k_fill_u32<<<blk(NH,256),256,0,stream>>>((unsigned*)hnbuf, 0u, NH);
    k_edge_max   <<<blk(EW,256),256,0,stream>>>(src, dst, qbuf, knbuf, kebuf, (unsigned*)mbuf);
    k_decode_max <<<blk(NH,256),256,0,stream>>>((unsigned*)mbuf, NH);
    k_edge_expsum<<<blk(EW,256),256,0,stream>>>(src, dst, qbuf, knbuf, kebuf, mbuf, ebuf, sbuf);
    k_edge_aggr  <<<blk(EW,256),256,0,stream>>>(src, dst, ebuf, sbuf, vnbuf, vebuf, hnbuf);

    // ------- h = LN([h_n, kind, ntype] @ Ww + Wb) -------
    k_cast_bf16<<<blk(NH,256),256,0,stream>>>(hnbuf, aggb, NH);
    k_wmma_gemm<<<gN,256,0,stream>>>(seg3(aggb,128,0,128, kindb,64,128,64, ntypeb,64,192,64),
                                     Wwt, 256, Wb, preln, N_);
    k_layernorm<<<blk(N_,8),256,0,stream>>>(preln, lng, lnb, hbuf, hb, N_);

    // ------- layer 2 projections (K2w/V2w rows: kind 0:64, ntype 64:128, eh 128:192, h 192:320) -------
    k_wmma_gemm<<<gN,256,0,stream>>>(seg3(kindb,64,0,64, ntypeb,64,64,64, hb,128,128,128),
                                     Q2wt, 256, Q2b, qbuf, N_);
    k_wmma_gemm<<<gN,256,0,stream>>>(seg3(kindb,64,0,64, ntypeb,64,64,64, hb,128,192,128),
                                     K2wt, 320, nullptr, knbuf, N_);
    k_wmma_gemm<<<gN,256,0,stream>>>(seg3(kindb,64,0,64, ntypeb,64,64,64, hb,128,192,128),
                                     V2wt, 320, nullptr, vnbuf, N_);
    k_wmma_gemm<<<gE,256,0,stream>>>(seg1(ehb,64,128,64), K2wt, 320, K2b, kebuf, E_);
    k_wmma_gemm<<<gE,256,0,stream>>>(seg1(ehb,64,128,64), V2wt, 320, V2b, vebuf, E_);

    // ------- layer 2 edge softmax + aggregate -------
    k_fill_u32<<<blk(NH,256),256,0,stream>>>((unsigned*)mbuf, ORD_NEG_INF, NH);
    k_fill_u32<<<blk(NH,256),256,0,stream>>>((unsigned*)sbuf, 0u, NH);
    k_fill_u32<<<blk(NH,256),256,0,stream>>>((unsigned*)hnbuf, 0u, NH);
    k_edge_max   <<<blk(EW,256),256,0,stream>>>(src, dst, qbuf, knbuf, kebuf, (unsigned*)mbuf);
    k_decode_max <<<blk(NH,256),256,0,stream>>>((unsigned*)mbuf, NH);
    k_edge_expsum<<<blk(EW,256),256,0,stream>>>(src, dst, qbuf, knbuf, kebuf, mbuf, ebuf, sbuf);
    k_edge_aggr  <<<blk(EW,256),256,0,stream>>>(src, dst, ebuf, sbuf, vnbuf, vebuf, hnbuf);

    // ------- h1 = LN([h_n1, h, kind, ntype] @ W2w + W2b) -------
    k_cast_bf16<<<blk(NH,256),256,0,stream>>>(hnbuf, aggb, NH);
    k_wmma_gemm<<<gN,256,0,stream>>>(seg4(aggb,128,0,128, hb,128,128,128,
                                          kindb,64,256,64, ntypeb,64,320,64),
                                     W2wt, 384, W2b, preln, N_);
    k_layernorm<<<blk(N_,8),256,0,stream>>>(preln, lng, lnb, h1buf, (__bf16*)nullptr, N_);

    // ------- global attention pooling readout -------
    k_fill_u32<<<1,256,0,stream>>>((unsigned*)gmbuf, ORD_NEG_INF, G_);
    k_fill_u32<<<1,256,0,stream>>>((unsigned*)gsbuf, 0u, G_);
    k_fill_u32<<<blk((long)G_*H_,256),256,0,stream>>>((unsigned*)out, 0u, (long)G_*H_);
    k_gate_logit<<<blk(N_,8),256,0,stream>>>(h1buf, gw, gb, gid, logitb, (unsigned*)gmbuf, N_);
    k_decode_max<<<1,256,0,stream>>>((unsigned*)gmbuf, G_);
    k_gate_exp  <<<blk(N_,256),256,0,stream>>>(logitb, gmbuf, gid, gebuf, gsbuf, N_);
    k_gate_out  <<<blk(N_,8),256,0,stream>>>(h1buf, gebuf, gsbuf, gid, out, N_);
}
